// CrystalDiffusionModel_48713519071926
// MI455X (gfx1250) — matrix operand order repaired and checked by reading, not compiled, not run on hardware
//
#include <hip/hip_runtime.h>

// ---------------------------------------------------------------------------
// CrystalDiffusionModel collapsed forward for gfx1250 (MI455X).
// Exact simplification: softmax over a single context token == 1, and the
// attention output overwrites `hu`, so the whole GNN conv is dead code.
// h_final = x@Wn + bn + 6*(cond@Wv@Wo + bo + silu(te))[batch]
// Heads computed with v_wmma_f32_16x16x32_f16 (f32 accumulate).
// silu uses v_rcp_f32 (fast reciprocal) instead of the IEEE divide expansion.
// ---------------------------------------------------------------------------

typedef __attribute__((ext_vector_type(16))) _Float16 v16h;
typedef __attribute__((ext_vector_type(8)))  float    v8f;
typedef __attribute__((ext_vector_type(4)))  unsigned int u32x4;

#define NNODES 20000
#define NGRAPH 128
#define HDIM   128

__device__ __forceinline__ float silu_f(float v) {
  // x * sigmoid(x); fast v_rcp_f32 instead of full-precision divide chain
  return v * __builtin_amdgcn_rcpf(1.0f + __expf(-v));
}

// ---------------------------------------------------------------------------
// Kernel 0: all per-graph work (time MLP, condition MLPs, attn path folded)
// graphC[g][c] = (cond[g] @ Wv @ Wo)[c] + bo[c] + silu(te[g])[c]
// ---------------------------------------------------------------------------
__global__ __launch_bounds__(128) void graph_kernel(
    const float* __restrict__ t,
    const float* __restrict__ topo, const float* __restrict__ stab, const float* __restrict__ sust,
    const float* __restrict__ tmW1, const float* __restrict__ tmb1,
    const float* __restrict__ tmW2, const float* __restrict__ tmb2,
    const float* __restrict__ tpW1, const float* __restrict__ tpb1,
    const float* __restrict__ tpW2, const float* __restrict__ tpb2,
    const float* __restrict__ sbW1, const float* __restrict__ sbb1,
    const float* __restrict__ sbW2, const float* __restrict__ sbb2,
    const float* __restrict__ ssW1, const float* __restrict__ ssb1,
    const float* __restrict__ ssW2, const float* __restrict__ ssb2,
    const float* __restrict__ cbW1, const float* __restrict__ cbb1,
    const float* __restrict__ cbW2, const float* __restrict__ cbb2,
    const float* __restrict__ Wv,  const float* __restrict__ Wo, const float* __restrict__ bo,
    float* __restrict__ graphC)
{
  const int g = blockIdx.x;
  const int j = threadIdx.x;

  __shared__ float te0[128];
  __shared__ float hid[256];
  __shared__ float st[128];
  __shared__ float tmp[64];
  __shared__ float cat[64];
  __shared__ float cnd[64];
  __shared__ float gv[128];

  const float tg = t[g];

  // sinusoidal embedding [sin(t*f), cos(t*f)], f_i = exp(-i*ln(1e4)/63)
  {
    int i = (j < 64) ? j : (j - 64);
    float f = __expf(-(float)i * (9.210340371976184f / 63.0f));
    float a = tg * f;
    te0[j] = (j < 64) ? __sinf(a) : __cosf(a);
  }
  __syncthreads();

  // time MLP layer 1: [128]->[256], silu
  for (int cc = 0; cc < 2; ++cc) {
    int c = j + cc * 128;
    float acc = tmb1[c];
    for (int k = 0; k < 128; ++k) acc += te0[k] * tmW1[k * 256 + c];
    hid[c] = silu_f(acc);
  }
  __syncthreads();

  // time MLP layer 2: [256]->[128]; st = silu(te)  (== ti per graph)
  {
    float acc = tmb2[j];
    for (int k = 0; k < 256; ++k) acc += hid[k] * tmW2[k * 128 + j];
    st[j] = silu_f(acc);
  }
  __syncthreads();

  // condition encoders in parallel: topo(32) | stab(16) | sust(16)
  if (j < 32) {
    float a = tpb1[j];
    for (int k = 0; k < 7; ++k) a += topo[g * 7 + k] * tpW1[k * 32 + j];
    tmp[j] = silu_f(a);
  } else if (j < 48) {
    int q = j - 32;
    float a = sbb1[q];
    for (int k = 0; k < 2; ++k) a += stab[g * 2 + k] * sbW1[k * 16 + q];
    tmp[j] = silu_f(a);
  } else if (j < 64) {
    int q = j - 48;
    float a = ssb1[q];
    for (int k = 0; k < 3; ++k) a += sust[g * 3 + k] * ssW1[k * 16 + q];
    tmp[j] = silu_f(a);
  }
  __syncthreads();
  if (j < 32) {
    float a = tpb2[j];
    for (int k = 0; k < 32; ++k) a += tmp[k] * tpW2[k * 32 + j];
    cat[j] = a;
  } else if (j < 48) {
    int q = j - 32;
    float a = sbb2[q];
    for (int k = 0; k < 16; ++k) a += tmp[32 + k] * sbW2[k * 16 + q];
    cat[j] = a;
  } else if (j < 64) {
    int q = j - 48;
    float a = ssb2[q];
    for (int k = 0; k < 16; ++k) a += tmp[48 + k] * ssW2[k * 16 + q];
    cat[j] = a;
  }
  __syncthreads();

  // combiner MLP: [64]->[64] silu ->[64]
  if (j < 64) {
    float a = cbb1[j];
    for (int k = 0; k < 64; ++k) a += cat[k] * cbW1[k * 64 + j];
    tmp[j] = silu_f(a);
  }
  __syncthreads();
  if (j < 64) {
    float a = cbb2[j];
    for (int k = 0; k < 64; ++k) a += tmp[k] * cbW2[k * 64 + j];
    cnd[j] = a;
  }
  __syncthreads();

  // gv = cond @ Wv   [64]->[128]
  {
    float a = 0.0f;
    for (int k = 0; k < 64; ++k) a += cnd[k] * Wv[k * 128 + j];
    gv[j] = a;
  }
  __syncthreads();

  // go = gv @ Wo + bo ; graphC = go + silu(te)
  {
    float a = bo[j];
    for (int k = 0; k < 128; ++k) a += gv[k] * Wo[k * 128 + j];
    graphC[g * 128 + j] = a + st[j];
  }
}

// ---------------------------------------------------------------------------
// Kernel 1: pack a row-major f32 weight [K][Nreal] into WMMA-B f16 layout.
// B operand of v_wmma_f32_16x16x32_f16: lane l holds col nt*16+(l&15),
// element i holds K = kc*32 + 16*(i>=8) + 8*(l>=16) + (i&7). Packed so the hot kernel does
// one contiguous 32B load per lane per (nt,kc) tile.
// ---------------------------------------------------------------------------
__global__ void pack_kernel(const float* __restrict__ src, _Float16* __restrict__ dst,
                            int K, int Nreal, int NT, int KC)
{
  int total = NT * KC * 512;
  for (int e = blockIdx.x * blockDim.x + threadIdx.x; e < total;
       e += gridDim.x * blockDim.x) {
    int i    = e & 15;
    int l    = (e >> 4) & 31;
    int tile = e >> 9;
    int kc   = tile % KC;
    int nt   = tile / KC;
    int k    = kc * 32 + ((i >> 3) << 4) + ((l >> 4) << 3) + (i & 7);
    int col  = nt * 16 + (l & 15);
    float v  = (col < Nreal && k < K) ? src[k * Nreal + col] : 0.0f;
    dst[e]   = (_Float16)v;
  }
}

// ---------------------------------------------------------------------------
// A-operand fetch from a row-major f16 LDS tile [16][stride].
// Layout (ISA 7.12.2, 16-bit A 16x32): row = l&15,
//   elems 0..7  -> K = k0 + 8*(l>=16) + 0..7
//   elems 8..15 -> K = k0 + 16 + 8*(l>=16) + 0..7
// -> two ds_load_b128 per lane.
// ---------------------------------------------------------------------------
__device__ __forceinline__ v16h load_a(const _Float16* base, int stride, int k0, int lane)
{
  const _Float16* p = base + (lane & 15) * stride + k0 + ((lane >> 4) << 3);
  union { v16h v; u32x4 u[2]; } a;
  a.u[0] = *(const u32x4*)(p);
  a.u[1] = *(const u32x4*)(p + 16);
  return a.v;
}

// ---------------------------------------------------------------------------
// Kernel 2: per-node pipeline. 4 waves/block, 16 nodes per wave.
// ---------------------------------------------------------------------------
__global__ __launch_bounds__(128) void node_kernel(
    const float* __restrict__ x, const int* __restrict__ batch,
    const float* __restrict__ Wn, const float* __restrict__ bn,
    const float* __restrict__ npb1, const float* __restrict__ npb2,
    const float* __restrict__ ppb1, const float* __restrict__ ppb2,
    const float* __restrict__ graphC,
    const _Float16* __restrict__ np1p, const _Float16* __restrict__ np2p,
    const _Float16* __restrict__ pp1p, const _Float16* __restrict__ pp2p,
    float* __restrict__ node_out, float* __restrict__ pos_out, int nNodes)
{
  __shared__ __align__(16) float    s_wn[12 * 128];
  __shared__ __align__(16) _Float16 s_h[4][16 * 128];
  __shared__ __align__(16) _Float16 s_hid[4][16 * 256];
  __shared__ __align__(16) float    s_x[4][16 * 12];
  __shared__ int                    s_gb[4][16];

  const int tid  = threadIdx.x;
  const int lane = tid & 31;
  const int wave = tid >> 5;
  const int tileBase = blockIdx.x * 64 + wave * 16;

  // stage Wn (block-wide) and x rows / batch ids (per wave)
  for (int idx = tid; idx < 12 * 128; idx += 128) s_wn[idx] = Wn[idx];
  if (lane < 16) {
    int node = tileBase + lane;
    if (node >= nNodes) node = nNodes - 1;
    for (int k = 0; k < 12; ++k) s_x[wave][lane * 12 + k] = x[node * 12 + k];
    s_gb[wave][lane] = batch[node];
  }
  __syncthreads();

  // h = x@Wn + bn + 6*graphC[batch]  -> f16 LDS tile [16][128]
  {
    const int c0 = lane * 4;
    const float b0 = bn[c0], b1 = bn[c0 + 1], b2 = bn[c0 + 2], b3 = bn[c0 + 3];
    for (int r = 0; r < 16; ++r) {
      const float* gc = graphC + s_gb[wave][r] * 128 + c0;
      float a0 = b0 + 6.0f * gc[0];
      float a1 = b1 + 6.0f * gc[1];
      float a2 = b2 + 6.0f * gc[2];
      float a3 = b3 + 6.0f * gc[3];
      for (int k = 0; k < 12; ++k) {
        float xv = s_x[wave][r * 12 + k];
        const float* w = s_wn + k * 128 + c0;
        a0 += xv * w[0]; a1 += xv * w[1]; a2 += xv * w[2]; a3 += xv * w[3];
      }
      _Float16* hp = &s_h[wave][r * 128 + c0];
      hp[0] = (_Float16)a0; hp[1] = (_Float16)a1;
      hp[2] = (_Float16)a2; hp[3] = (_Float16)a3;
    }
  }
  __syncthreads();

  const int colL   = lane & 15;
  const int rowOff = (lane >> 4) << 3;  // 0 or 8

  // G1: hid1 = silu(h @ np_W1 + np_b1)   [16x128]@[128x256]
  for (int nt = 0; nt < 16; ++nt) {
    v8f acc = {};
    for (int kc = 0; kc < 4; ++kc) {
      v16h a = load_a(s_h[wave], 128, kc * 32, lane);
      v16h b = *(const v16h*)(np1p + (nt * 4 + kc) * 512 + lane * 16);
      acc = __builtin_amdgcn_wmma_f32_16x16x32_f16(false, a, false, b, (short)0, acc,
                                                   false, false);
    }
    int col = nt * 16 + colL;
    float bias = npb1[col];
    for (int r = 0; r < 8; ++r) {
      float v = silu_f(acc[r] + bias);
      s_hid[wave][(r + rowOff) * 256 + col] = (_Float16)v;
    }
  }
  __syncthreads();

  // G2: node_pred = hid1 @ np_W2 + np_b2   [16x256]@[256x12->16]
  {
    v8f acc = {};
    for (int kc = 0; kc < 8; ++kc) {
      v16h a = load_a(s_hid[wave], 256, kc * 32, lane);
      v16h b = *(const v16h*)(np2p + kc * 512 + lane * 16);
      acc = __builtin_amdgcn_wmma_f32_16x16x32_f16(false, a, false, b, (short)0, acc,
                                                   false, false);
    }
    if (colL < 12) {
      float bias = npb2[colL];
      for (int r = 0; r < 8; ++r) {
        int node = tileBase + r + rowOff;
        if (node < nNodes) node_out[node * 12 + colL] = acc[r] + bias;
      }
    }
  }
  __syncthreads();

  // G3: hid2 = silu(h @ pp_W1 + pp_b1)   [16x128]@[128x128] (reuse s_hid)
  for (int nt = 0; nt < 8; ++nt) {
    v8f acc = {};
    for (int kc = 0; kc < 4; ++kc) {
      v16h a = load_a(s_h[wave], 128, kc * 32, lane);
      v16h b = *(const v16h*)(pp1p + (nt * 4 + kc) * 512 + lane * 16);
      acc = __builtin_amdgcn_wmma_f32_16x16x32_f16(false, a, false, b, (short)0, acc,
                                                   false, false);
    }
    int col = nt * 16 + colL;
    float bias = ppb1[col];
    for (int r = 0; r < 8; ++r) {
      float v = silu_f(acc[r] + bias);
      s_hid[wave][(r + rowOff) * 128 + col] = (_Float16)v;
    }
  }
  __syncthreads();

  // G4: pos_pred = hid2 @ pp_W2 + pp_b2   [16x128]@[128x3->16]
  {
    v8f acc = {};
    for (int kc = 0; kc < 4; ++kc) {
      v16h a = load_a(s_hid[wave], 128, kc * 32, lane);
      v16h b = *(const v16h*)(pp2p + kc * 512 + lane * 16);
      acc = __builtin_amdgcn_wmma_f32_16x16x32_f16(false, a, false, b, (short)0, acc,
                                                   false, false);
    }
    if (colL < 3) {
      float bias = ppb2[colL];
      for (int r = 0; r < 8; ++r) {
        int node = tileBase + r + rowOff;
        if (node < nNodes) pos_out[node * 3 + colL] = acc[r] + bias;
      }
    }
  }
}

// ---------------------------------------------------------------------------
extern "C" void kernel_launch(void* const* d_in, const int* in_sizes, int n_in,
                              void* d_out, int out_size, void* d_ws, size_t ws_size,
                              hipStream_t stream)
{
  (void)in_sizes; (void)n_in; (void)out_size; (void)ws_size;

  const float* x     = (const float*)d_in[0];
  const float* t     = (const float*)d_in[4];
  const float* topo  = (const float*)d_in[5];
  const float* stab  = (const float*)d_in[6];
  const float* sust  = (const float*)d_in[7];
  const int*   batch = (const int*)d_in[8];
  const float* Wn    = (const float*)d_in[9];
  const float* bn    = (const float*)d_in[10];
  const float* tmW1  = (const float*)d_in[13];
  const float* tmb1  = (const float*)d_in[14];
  const float* tmW2  = (const float*)d_in[15];
  const float* tmb2  = (const float*)d_in[16];
  const float* tpW1  = (const float*)d_in[17];
  const float* tpb1  = (const float*)d_in[18];
  const float* tpW2  = (const float*)d_in[19];
  const float* tpb2  = (const float*)d_in[20];
  const float* sbW1  = (const float*)d_in[21];
  const float* sbb1  = (const float*)d_in[22];
  const float* sbW2  = (const float*)d_in[23];
  const float* sbb2  = (const float*)d_in[24];
  const float* ssW1  = (const float*)d_in[25];
  const float* ssb1  = (const float*)d_in[26];
  const float* ssW2  = (const float*)d_in[27];
  const float* ssb2  = (const float*)d_in[28];
  const float* cbW1  = (const float*)d_in[29];
  const float* cbb1  = (const float*)d_in[30];
  const float* cbW2  = (const float*)d_in[31];
  const float* cbb2  = (const float*)d_in[32];
  const float* Wv    = (const float*)d_in[35];
  const float* Wo    = (const float*)d_in[36];
  const float* bo    = (const float*)d_in[37];
  const float* npW1  = (const float*)d_in[44];
  const float* npb1  = (const float*)d_in[45];
  const float* npW2  = (const float*)d_in[46];
  const float* npb2  = (const float*)d_in[47];
  const float* ppW1  = (const float*)d_in[48];
  const float* ppb1  = (const float*)d_in[49];
  const float* ppW2  = (const float*)d_in[50];
  const float* ppb2  = (const float*)d_in[51];

  char* ws = (char*)d_ws;
  float*    graphC = (float*)(ws);                 //  65536 B : [128][128] f32
  _Float16* np1p   = (_Float16*)(ws + 65536);      //  65536 B : 16nt x 4kc x 512
  _Float16* np2p   = (_Float16*)(ws + 131072);     //   8192 B : 1nt x 8kc x 512
  _Float16* pp1p   = (_Float16*)(ws + 139264);     //  32768 B : 8nt x 4kc x 512
  _Float16* pp2p   = (_Float16*)(ws + 172032);     //   4096 B : 1nt x 4kc x 512

  float* node_out = (float*)d_out;                 // [20000][12]
  float* pos_out  = node_out + NNODES * 12;        // [20000][3]

  graph_kernel<<<NGRAPH, 128, 0, stream>>>(
      t, topo, stab, sust,
      tmW1, tmb1, tmW2, tmb2,
      tpW1, tpb1, tpW2, tpb2,
      sbW1, sbb1, sbW2, sbb2,
      ssW1, ssb1, ssW2, ssb2,
      cbW1, cbb1, cbW2, cbb2,
      Wv, Wo, bo, graphC);

  pack_kernel<<<(16 * 4 * 512 + 255) / 256, 256, 0, stream>>>(npW1, np1p, 128, 256, 16, 4);
  pack_kernel<<<(1 * 8 * 512 + 255) / 256, 256, 0, stream>>>(npW2, np2p, 256, 12, 1, 8);
  pack_kernel<<<(8 * 4 * 512 + 255) / 256, 256, 0, stream>>>(ppW1, pp1p, 128, 128, 8, 4);
  pack_kernel<<<(1 * 4 * 512 + 255) / 256, 256, 0, stream>>>(ppW2, pp2p, 128, 3, 1, 4);

  node_kernel<<<(NNODES + 63) / 64, 128, 0, stream>>>(
      x, batch, Wn, bn, npb1, npb2, ppb1, ppb2, graphC,
      np1p, np2p, pp1p, pp2p, node_out, pos_out, NNODES);
}
